// MoDeDiT_16071767622268
// MI455X (gfx1250) — matrix operand order repaired and checked by reading, not compiled, not run on hardware
//
#include <hip/hip_runtime.h>
#include <math.h>

// ---------------- CDNA5 WMMA types ----------------
typedef __attribute__((ext_vector_type(16))) __bf16 v16bf;
typedef __attribute__((ext_vector_type(8)))  __bf16 v8bf;
typedef __attribute__((ext_vector_type(8)))  float  v8f;

#define DEV __device__ __forceinline__

DEV float gelu_erf(float x) { return 0.5f * x * (1.0f + erff(x * 0.70710678118654752f)); }
DEV int imin(int a, int b) { return a < b ? a : b; }

// ---- fragment loaders (contiguous-k layouts: k = half*16 + s, wave32), 128-bit LDS loads except frag_b ----
// A 16x32 bf16: lane = half*16 + m ; per-lane data = row[k0+half*16 .. +15]
DEV v16bf frag_a(const __bf16* __restrict__ p, int ld, int m0, int k0, int lane) {
    int m = m0 + (lane & 15), half = lane >> 4;
    const __bf16* r = p + (size_t)m * ld + k0 + half * 16;
    v8bf lo = *(const v8bf*)(r);
    v8bf hi = *(const v8bf*)(r + 8);
    return __builtin_shufflevector(lo, hi, 0, 1, 2, 3, 4, 5, 6, 7,
                                           8, 9, 10, 11, 12, 13, 14, 15);
}
// B 32x16 bf16 from an n-major tile: element (k,n) = p[(n0+n)*ld + k0 + k];
// per-lane k = half*16 + 0..15  -> contiguous 32 bytes
DEV v16bf frag_bt(const __bf16* __restrict__ p, int ld, int k0, int n0, int lane) {
    int n = n0 + (lane & 15), half = lane >> 4;
    const __bf16* r = p + (size_t)n * ld + k0 + half * 16;
    v8bf lo = *(const v8bf*)(r);
    v8bf hi = *(const v8bf*)(r + 8);
    return __builtin_shufflevector(lo, hi, 0, 1, 2, 3, 4, 5, 6, 7,
                                           8, 9, 10, 11, 12, 13, 14, 15);
}
// B 32x16 bf16 from a k-major tile (strided; only used for attention V)
DEV v16bf frag_b(const __bf16* __restrict__ p, int ld, int k0, int n0, int lane) {
    int n = n0 + (lane & 15), half = lane >> 4;
    const __bf16* c = p + (size_t)(k0 + half * 16) * ld + n;
    v16bf b;
#pragma unroll
    for (int v = 0; v < 8; ++v) {
        b[2 * v]     = c[(2 * v) * ld];
        b[2 * v + 1] = c[(2 * v + 1) * ld];
    }
    return b;
}

// ---------------- small utility kernels ----------------
__global__ void cvt_k(const float* __restrict__ s, __bf16* __restrict__ d, int n) {
    int i = blockIdx.x * 256 + threadIdx.x;
    if (i < n) d[i] = (__bf16)s[i];
}

// f32 [R][C] -> bf16 transposed [C][R], 32x32 tiles via LDS
__global__ __launch_bounds__(256) void cvt_t_k(const float* __restrict__ s,
                                               __bf16* __restrict__ d, int R, int C) {
    __shared__ __bf16 tile[32][33];
    int c0 = blockIdx.x * 32, r0 = blockIdx.y * 32;
    int tx = threadIdx.x & 31, ty = threadIdx.x >> 5;   // 32 x 8
#pragma unroll
    for (int i = 0; i < 32; i += 8)
        tile[ty + i][tx] = (__bf16)s[(size_t)(r0 + ty + i) * C + c0 + tx];
    __syncthreads();
#pragma unroll
    for (int i = 0; i < 32; i += 8)
        d[(size_t)(c0 + ty + i) * R + r0 + tx] = tile[tx][ty + i];
}

__global__ void zero_cnt_k(int* cnt) {
    if (threadIdx.x < 4) cnt[threadIdx.x] = 0;
}

// LayerNorm: one block per token row of 1024, f32 in -> bf16 out
__global__ __launch_bounds__(256) void ln_k(const float* __restrict__ x,
                                            const float* __restrict__ g,
                                            const float* __restrict__ bb,
                                            __bf16* __restrict__ out) {
    int t = blockIdx.x;
    const float* row = x + (size_t)t * 1024;
    float vals[4], s = 0.f, s2 = 0.f;
#pragma unroll
    for (int i = 0; i < 4; ++i) {
        float v = row[threadIdx.x + i * 256];
        vals[i] = v; s += v; s2 += v * v;
    }
#pragma unroll
    for (int d = 1; d < 32; d <<= 1) { s += __shfl_xor(s, d, 32); s2 += __shfl_xor(s2, d, 32); }
    __shared__ float sh[16];
    int wid = threadIdx.x >> 5, lane = threadIdx.x & 31;
    if (lane == 0) { sh[wid] = s; sh[8 + wid] = s2; }
    __syncthreads();
    if (threadIdx.x == 0) {
        float ts = 0.f, ts2 = 0.f;
        for (int w = 0; w < 8; ++w) { ts += sh[w]; ts2 += sh[8 + w]; }
        float mu = ts * (1.0f / 1024.0f);
        float var = ts2 * (1.0f / 1024.0f) - mu * mu;
        sh[0] = mu; sh[1] = rsqrtf(var + 1e-5f);
    }
    __syncthreads();
    float mu = sh[0], rs = sh[1];
#pragma unroll
    for (int i = 0; i < 4; ++i) {
        int k = threadIdx.x + i * 256;
        out[(size_t)t * 1024 + k] = (__bf16)((vals[i] - mu) * rs * g[k] + bb[k]);
    }
}

// Build router input [T,2048] = [h2 | bf16(cond broadcast over seq)]
__global__ void rin_k(const __bf16* __restrict__ h2, const float* __restrict__ cond,
                      __bf16* __restrict__ rin) {
    size_t i = (size_t)blockIdx.x * 256 + threadIdx.x;   // T*2048 elements
    int t = (int)(i >> 11), k = (int)(i & 2047);
    if (k < 1024) rin[i] = h2[(size_t)t * 1024 + k];
    else          rin[i] = (__bf16)cond[(size_t)(t >> 8) * 1024 + (k - 1024)];
}

// ---------------- generic bf16 WMMA GEMM (128x64 block tile, double-buffered) ----
// MODE 0: f32 store (+RES residual), MODE 1: bf16 store, MODE 2: scatter f32 accum
// BwT is the TRANSPOSED weight: [N][K] bf16.
template <int ACT, int MODE, int RES>
__global__ __launch_bounds__(256) void gemm_bf16_k(
    const __bf16* __restrict__ A, const __bf16* __restrict__ BwT,
    const float* __restrict__ bias, const float* __restrict__ resid,
    void* __restrict__ outp, int M, int N, int K,
    const int* __restrict__ gather, const int* __restrict__ cntPtr,
    const int* __restrict__ list, const float* __restrict__ wgt) {
    __shared__ __bf16 As[2][128][40];
    __shared__ __bf16 Bs[2][64][40];
    const int tid = threadIdx.x, lane = tid & 31, wid = tid >> 5;
    const int wm = wid & 3, wn = wid >> 2;   // wave tile: 32 rows x 32 cols
    const int n0 = blockIdx.x * 64, m0 = blockIdx.y * 128;

    int cnt = M;
    if (cntPtr) { cnt = *cntPtr; if (m0 >= cnt) return; }

    // A tile 128x32: each thread 16 elems (2x uint4)
    const int ar = tid >> 1, ac = (tid & 1) * 16;
    int arow = m0 + ar;
    int asrc = gather ? gather[imin(arow, cnt - 1)] : imin(arow, cnt - 1);
    const __bf16* aptr = A + (size_t)asrc * K + ac;
    // B tile 64x32 (n-major): each thread 8 elems (1x uint4)
    const int bn = tid >> 2, bk = (tid & 3) * 8;
    const __bf16* bptr = BwT + (size_t)(n0 + bn) * K + bk;

    // prologue: stage tile 0
    {
        uint4 a0r = *(const uint4*)(aptr);
        uint4 a1r = *(const uint4*)(aptr + 8);
        uint4 b0r = *(const uint4*)(bptr);
        *(uint4*)&As[0][ar][ac]     = a0r;
        *(uint4*)&As[0][ar][ac + 8] = a1r;
        *(uint4*)&Bs[0][bn][bk]     = b0r;
    }
    __syncthreads();

    v8f c00 = {}, c01 = {}, c10 = {}, c11 = {};
    int cur = 0;
    for (int k0 = 0; k0 < K; k0 += 32) {
        const bool more = (k0 + 32) < K;
        uint4 an0, an1, bn0;
        if (more) {                      // prefetch next tile into regs
            an0 = *(const uint4*)(aptr + k0 + 32);
            an1 = *(const uint4*)(aptr + k0 + 40);
            bn0 = *(const uint4*)(bptr + k0 + 32);
        }
        v16bf a0 = frag_a(&As[cur][0][0], 40, wm * 32, 0, lane);
        v16bf a1 = frag_a(&As[cur][0][0], 40, wm * 32 + 16, 0, lane);
        v16bf b0 = frag_bt(&Bs[cur][0][0], 40, 0, wn * 32, lane);
        v16bf b1 = frag_bt(&Bs[cur][0][0], 40, 0, wn * 32 + 16, lane);
        c00 = __builtin_amdgcn_wmma_f32_16x16x32_bf16(false, a0, false, b0, (short)0, c00, false, false);
        c01 = __builtin_amdgcn_wmma_f32_16x16x32_bf16(false, a0, false, b1, (short)0, c01, false, false);
        c10 = __builtin_amdgcn_wmma_f32_16x16x32_bf16(false, a1, false, b0, (short)0, c10, false, false);
        c11 = __builtin_amdgcn_wmma_f32_16x16x32_bf16(false, a1, false, b1, (short)0, c11, false, false);
        if (more) {                      // stage into the other buffer
            *(uint4*)&As[cur ^ 1][ar][ac]     = an0;
            *(uint4*)&As[cur ^ 1][ar][ac + 8] = an1;
            *(uint4*)&Bs[cur ^ 1][bn][bk]     = bn0;
        }
        __syncthreads();
        cur ^= 1;
    }

    const int half = lane >> 4, cn = lane & 15;
#pragma unroll
    for (int mi = 0; mi < 2; ++mi) {
#pragma unroll
        for (int r = 0; r < 8; ++r) {
            int row = m0 + wm * 32 + mi * 16 + r + 8 * half;
            if (row >= cnt) continue;
#pragma unroll
            for (int ni = 0; ni < 2; ++ni) {
                int col = n0 + wn * 32 + ni * 16 + cn;
                float v = mi == 0 ? (ni == 0 ? c00[r] : c01[r])
                                  : (ni == 0 ? c10[r] : c11[r]);
                if (bias) v += bias[col];
                if (ACT == 1) v = gelu_erf(v);
                if (MODE == 2) {
                    int t = list[row];
                    float w = wgt[row];
                    ((float*)outp)[(size_t)t * N + col] += w * v;  // sequential expert launches: race-free
                } else if (MODE == 1) {
                    ((__bf16*)outp)[(size_t)row * N + col] = (__bf16)v;
                } else {
                    if (RES) v += resid[(size_t)row * N + col];
                    ((float*)outp)[(size_t)row * N + col] = v;
                }
            }
        }
    }
}

// ---------------- flash attention (per b,h, 64-query tile) ----------------
__global__ __launch_bounds__(128) void attn_flash_k(const __bf16* __restrict__ qkv,
                                                    __bf16* __restrict__ o) {
    const int SEQ = 256, S3 = 3072;
    int bid = blockIdx.x;
    int qt = bid & 3, h = (bid >> 2) & 15, b = bid >> 6;
    const int tid = threadIdx.x, lane = tid & 31, wid = tid >> 5;

    __shared__ __bf16 Qs[64][72];
    __shared__ __bf16 Ks[64][72];
    __shared__ __bf16 Vs[64][72];
    __shared__ __bf16 Ps[4][16][72];

    {   // load Q tile (64 queries x 64 head dims)
        int r = tid >> 1, c0l = (tid & 1) * 32;
        const __bf16* src = qkv + (size_t)(b * SEQ + qt * 64 + r) * S3 + h * 64 + c0l;
#pragma unroll
        for (int j = 0; j < 4; ++j)
            *(uint4*)&Qs[r][c0l + j * 8] = *(const uint4*)(src + j * 8);
    }
    __syncthreads();
    v16bf qa0 = frag_a(&Qs[0][0], 72, wid * 16, 0, lane);
    v16bf qa1 = frag_a(&Qs[0][0], 72, wid * 16, 32, lane);

    v8f s[4], oa[4];
#pragma unroll
    for (int i = 0; i < 4; ++i) oa[i] = (v8f){};
    float mrow[8], lrow[8];
#pragma unroll
    for (int r = 0; r < 8; ++r) { mrow[r] = -1e30f; lrow[r] = 0.f; }
    const float scale = 0.125f;   // HD^-0.5

    for (int kt = 0; kt < 4; ++kt) {
        __syncthreads();
        {   // load K,V tiles (64 keys x 64 dims)
            int r = tid >> 1, c0l = (tid & 1) * 32;
            const __bf16* ks = qkv + (size_t)(b * SEQ + kt * 64 + r) * S3 + 1024 + h * 64 + c0l;
            const __bf16* vs = qkv + (size_t)(b * SEQ + kt * 64 + r) * S3 + 2048 + h * 64 + c0l;
#pragma unroll
            for (int j = 0; j < 4; ++j) {
                *(uint4*)&Ks[r][c0l + j * 8] = *(const uint4*)(ks + j * 8);
                *(uint4*)&Vs[r][c0l + j * 8] = *(const uint4*)(vs + j * 8);
            }
        }
        __syncthreads();

        // S = Q * K^T  (16 queries x 64 keys per wave); Ks is key-major -> frag_bt wide
#pragma unroll
        for (int n = 0; n < 4; ++n) {
            s[n] = (v8f){};
            s[n] = __builtin_amdgcn_wmma_f32_16x16x32_bf16(false, qa0, false,
                        frag_bt(&Ks[0][0], 72, 0,  n * 16, lane), (short)0, s[n], false, false);
            s[n] = __builtin_amdgcn_wmma_f32_16x16x32_bf16(false, qa1, false,
                        frag_bt(&Ks[0][0], 72, 32, n * 16, lane), (short)0, s[n], false, false);
        }

        // online softmax per row (row = r + 8*half, 16 lanes share a row)
        const int half = lane >> 4, pc = lane & 15;
#pragma unroll
        for (int r = 0; r < 8; ++r) {
            float x0 = s[0][r] * scale, x1 = s[1][r] * scale;
            float x2 = s[2][r] * scale, x3 = s[3][r] * scale;
            float mx = fmaxf(fmaxf(x0, x1), fmaxf(x2, x3));
#pragma unroll
            for (int d = 1; d < 16; d <<= 1) mx = fmaxf(mx, __shfl_xor(mx, d, 32));
            float mnew = fmaxf(mrow[r], mx);
            float corr = __expf(mrow[r] - mnew);
            mrow[r] = mnew;
            float p0 = __expf(x0 - mnew), p1 = __expf(x1 - mnew);
            float p2 = __expf(x2 - mnew), p3 = __expf(x3 - mnew);
            float ls = p0 + p1 + p2 + p3;
#pragma unroll
            for (int d = 1; d < 16; d <<= 1) ls += __shfl_xor(ls, d, 32);
            lrow[r] = lrow[r] * corr + ls;
            oa[0][r] *= corr; oa[1][r] *= corr; oa[2][r] *= corr; oa[3][r] *= corr;
            int pr = r + 8 * half;
            Ps[wid][pr][0 + pc]  = (__bf16)p0;
            Ps[wid][pr][16 + pc] = (__bf16)p1;
            Ps[wid][pr][32 + pc] = (__bf16)p2;
            Ps[wid][pr][48 + pc] = (__bf16)p3;
        }

        // O += P * V  (P is row-major per-wave tile -> frag_a wide; V key-major -> strided frag_b)
        v16bf pa0 = frag_a(&Ps[wid][0][0], 72, 0, 0, lane);
        v16bf pa1 = frag_a(&Ps[wid][0][0], 72, 0, 32, lane);
#pragma unroll
        for (int n = 0; n < 4; ++n) {
            oa[n] = __builtin_amdgcn_wmma_f32_16x16x32_bf16(false, pa0, false,
                        frag_b(&Vs[0][0], 72, 0,  n * 16, lane), (short)0, oa[n], false, false);
            oa[n] = __builtin_amdgcn_wmma_f32_16x16x32_bf16(false, pa1, false,
                        frag_b(&Vs[0][0], 72, 32, n * 16, lane), (short)0, oa[n], false, false);
        }
    }

    const int half = lane >> 4, cn = lane & 15;
#pragma unroll
    for (int r = 0; r < 8; ++r) {
        int row = qt * 64 + wid * 16 + r + 8 * half;
        float inv = 1.0f / lrow[r];
        size_t base = (size_t)(b * SEQ + row) * 1024 + h * 64;
#pragma unroll
        for (int n = 0; n < 4; ++n)
            o[base + n * 16 + cn] = (__bf16)(oa[n][r] * inv);
    }
}

// ---------------- router: fc2 + softmax + top-2 + scatter ----------------
__global__ __launch_bounds__(256) void router_k(const __bf16* __restrict__ rh,
                                                const __bf16* __restrict__ w2,
                                                const float* __restrict__ b2,
                                                int* __restrict__ cnt,
                                                int* __restrict__ list,
                                                float* __restrict__ wgt) {
    int wid = threadIdx.x >> 5, lane = threadIdx.x & 31;
    int t = blockIdx.x * 8 + wid;
    const __bf16* row = rh + (size_t)t * 2048;
    float l0 = 0.f, l1 = 0.f, l2 = 0.f, l3 = 0.f;
    for (int k = lane; k < 2048; k += 32) {
        float a = (float)row[k];
        l0 += a * (float)w2[k * 4 + 0];
        l1 += a * (float)w2[k * 4 + 1];
        l2 += a * (float)w2[k * 4 + 2];
        l3 += a * (float)w2[k * 4 + 3];
    }
#pragma unroll
    for (int d = 1; d < 32; d <<= 1) {
        l0 += __shfl_xor(l0, d, 32); l1 += __shfl_xor(l1, d, 32);
        l2 += __shfl_xor(l2, d, 32); l3 += __shfl_xor(l3, d, 32);
    }
    if (lane == 0) {
        float p[4] = {l0 + b2[0], l1 + b2[1], l2 + b2[2], l3 + b2[3]};
        float mx = fmaxf(fmaxf(p[0], p[1]), fmaxf(p[2], p[3]));
        float sm = 0.f;
        for (int e = 0; e < 4; ++e) { p[e] = __expf(p[e] - mx); sm += p[e]; }
        for (int e = 0; e < 4; ++e) {
            p[e] /= sm;
            p[e] = fminf(fmaxf(p[e], 1e-9f), 1.0f - 1e-9f);
        }
        int i0 = 0;
        for (int e = 1; e < 4; ++e) if (p[e] > p[i0]) i0 = e;
        int i1 = -1;
        for (int e = 0; e < 4; ++e) if (e != i0 && (i1 < 0 || p[e] > p[i1])) i1 = e;
        float ws = p[i0] + p[i1];
        int a0 = atomicAdd(&cnt[i0], 1);
        list[i0 * 8192 + a0] = t; wgt[i0 * 8192 + a0] = p[i0] / ws;
        int a1 = atomicAdd(&cnt[i1], 1);
        list[i1 * 8192 + a1] = t; wgt[i1 * 8192 + a1] = p[i1] / ws;
    }
}

// ---------------- host wiring ----------------
extern "C" void kernel_launch(void* const* d_in, const int* in_sizes, int n_in,
                              void* d_out, int out_size, void* d_ws, size_t ws_size,
                              hipStream_t stream) {
    const int T = 8192;
    const float* x      = (const float*)d_in[0];
    const float* cond   = (const float*)d_in[1];
    const float* ln1_g  = (const float*)d_in[2];
    const float* ln1_b  = (const float*)d_in[3];
    const float* qkv_w  = (const float*)d_in[4];
    const float* proj_w = (const float*)d_in[5];
    const float* proj_b = (const float*)d_in[6];
    const float* ln2_g  = (const float*)d_in[7];
    const float* ln2_b  = (const float*)d_in[8];
    const float* r1_w   = (const float*)d_in[9];
    const float* r1_b   = (const float*)d_in[10];
    const float* r2_w   = (const float*)d_in[11];
    const float* r2_b   = (const float*)d_in[12];
    const float* e_w1   = (const float*)d_in[13];
    const float* e_b1   = (const float*)d_in[14];
    const float* e_w2   = (const float*)d_in[15];
    const float* e_b2   = (const float*)d_in[16];
    float* out = (float*)d_out;

    char* w = (char*)d_ws;
    auto alloc = [&](size_t bytes) -> void* {
        void* p = (void*)w;
        w += (bytes + 255) & ~(size_t)255;
        return p;
    };
    // transposed bf16 weights: [N][K]
    __bf16* wqkvT = (__bf16*)alloc((size_t)3072 * 1024 * 2);
    __bf16* wprojT = (__bf16*)alloc((size_t)1024 * 1024 * 2);
    __bf16* wr1T  = (__bf16*)alloc((size_t)2048 * 2048 * 2);
    __bf16* wr2   = (__bf16*)alloc((size_t)2048 * 4 * 2);          // kept [K][4]
    __bf16* we1T  = (__bf16*)alloc((size_t)4 * 4096 * 1024 * 2);
    __bf16* we2T  = (__bf16*)alloc((size_t)4 * 1024 * 4096 * 2);
    __bf16* h1   = (__bf16*)alloc((size_t)T * 1024 * 2);
    __bf16* qkvb = (__bf16*)alloc((size_t)T * 3072 * 2);
    __bf16* obuf = (__bf16*)alloc((size_t)T * 1024 * 2);
    __bf16* h2   = (__bf16*)alloc((size_t)T * 1024 * 2);
    __bf16* rin  = (__bf16*)alloc((size_t)T * 2048 * 2);
    __bf16* rh   = (__bf16*)alloc((size_t)T * 2048 * 2);
    __bf16* hid  = (__bf16*)alloc((size_t)T * 4096 * 2);
    int*   cnt  = (int*)alloc(64);
    int*   list = (int*)alloc((size_t)4 * T * 4);
    float* wgt  = (float*)alloc((size_t)4 * T * 4);

    auto cvtT = [&](const float* s, __bf16* d, int R, int C) {
        cvt_t_k<<<dim3(C / 32, R / 32), 256, 0, stream>>>(s, d, R, C);
    };
    cvtT(qkv_w, wqkvT, 1024, 3072);
    cvtT(proj_w, wprojT, 1024, 1024);
    cvtT(r1_w, wr1T, 2048, 2048);
    cvt_k<<<(2048 * 4 + 255) / 256, 256, 0, stream>>>(r2_w, wr2, 2048 * 4);
    for (int e = 0; e < 4; ++e) {
        cvtT(e_w1 + (size_t)e * 1024 * 4096, we1T + (size_t)e * 4096 * 1024, 1024, 4096);
        cvtT(e_w2 + (size_t)e * 4096 * 1024, we2T + (size_t)e * 1024 * 4096, 4096, 1024);
    }
    zero_cnt_k<<<1, 32, 0, stream>>>(cnt);

    // attention block
    ln_k<<<T, 256, 0, stream>>>(x, ln1_g, ln1_b, h1);
    gemm_bf16_k<0, 1, 0><<<dim3(48, 64), 256, 0, stream>>>(
        h1, wqkvT, nullptr, nullptr, qkvb, T, 3072, 1024, nullptr, nullptr, nullptr, nullptr);
    attn_flash_k<<<2048, 128, 0, stream>>>(qkvb, obuf);
    gemm_bf16_k<0, 0, 1><<<dim3(16, 64), 256, 0, stream>>>(
        obuf, wprojT, proj_b, x, out, T, 1024, 1024, nullptr, nullptr, nullptr, nullptr); // x1 -> d_out

    // router
    ln_k<<<T, 256, 0, stream>>>(out, ln2_g, ln2_b, h2);
    rin_k<<<(T * 2048) / 256, 256, 0, stream>>>(h2, cond, rin);
    gemm_bf16_k<1, 1, 0><<<dim3(32, 64), 256, 0, stream>>>(
        rin, wr1T, r1_b, nullptr, rh, T, 2048, 2048, nullptr, nullptr, nullptr, nullptr);
    router_k<<<T / 8, 256, 0, stream>>>(rh, wr2, r2_b, cnt, list, wgt);

    // experts (top-2 gathered; sequential launches per expert)
    for (int e = 0; e < 4; ++e) {
        gemm_bf16_k<1, 1, 0><<<dim3(64, 64), 256, 0, stream>>>(
            h2, we1T + (size_t)e * 4096 * 1024, e_b1 + e * 4096, nullptr, hid,
            T, 4096, 1024, list + e * T, cnt + e, nullptr, nullptr);
        gemm_bf16_k<0, 2, 0><<<dim3(16, 64), 256, 0, stream>>>(
            hid, we2T + (size_t)e * 1024 * 4096, e_b2 + e * 1024, nullptr, out,
            T, 1024, 4096, nullptr, cnt + e, list + e * T, wgt + e * T);
    }
    (void)in_sizes; (void)n_in; (void)out_size; (void)ws_size;
}
